// BitLinear_33165737459797
// MI455X (gfx1250) — compile-verified, matrix operand order checked by reading, not run on hardware
//
#include <hip/hip_runtime.h>
#include <hip/hip_bf16.h>

typedef __attribute__((ext_vector_type(8))) int v8i;
typedef __attribute__((ext_vector_type(2))) int v2i;

#define K_DIM   2048
#define M_DIM   8192   // 4 * 2048
#define N_DIM   8192
#define W_ELEMS (8192u * 2048u)   // 16,777,216
#define X_ELEMS (8192u * 2048u)

// ---------------- workspace layout ----------------
// [0,4)          : float  w_abs_sum
// [256, 33024)   : float  scales[8192]
// [65536, +16MB) : int8   x_q  (8192 x 2048)
// [65536+16MB,.) : int8   w_q  (8192 x 2048)

// ---------------------------------------------------------------------------
// Pass 0: sum(|weight|) -> ws[0]
// ---------------------------------------------------------------------------
__global__ __launch_bounds__(256) void wsum_kernel(const float* __restrict__ w,
                                                   float* __restrict__ wsum) {
    __shared__ float red[256];
    const int t = threadIdx.x;
    size_t i = (size_t)blockIdx.x * blockDim.x + t;
    const size_t stride = (size_t)gridDim.x * blockDim.x;
    float s = 0.0f;
    for (; i < W_ELEMS; i += stride) s += fabsf(w[i]);
    red[t] = s;
    __syncthreads();
    for (int off = 128; off > 0; off >>= 1) {
        if (t < off) red[t] += red[t + off];
        __syncthreads();
    }
    if (t == 0) atomicAdd(wsum, red[0]);
}

// ---------------------------------------------------------------------------
// Pass 1: ternary weight quantization -> int8 {-1,0,1}, row-major over K
// ---------------------------------------------------------------------------
__global__ __launch_bounds__(256) void quant_w_kernel(const float* __restrict__ w,
                                                      const float* __restrict__ wsum,
                                                      signed char* __restrict__ wq) {
    const float wscale = wsum[0] * (1.0f / 16777216.0f);
    const float thr = 0.5f * wscale;
    const size_t base = ((size_t)blockIdx.x * blockDim.x + threadIdx.x) * 8u;
    if (base >= W_ELEMS) return;
    const float4 a = *(const float4*)(w + base);
    const float4 b = *(const float4*)(w + base + 4);
    float v[8] = {a.x, a.y, a.z, a.w, b.x, b.y, b.z, b.w};
    int lo = 0, hi = 0;
#pragma unroll
    for (int i = 0; i < 4; ++i) {
        int q = (v[i] > thr) ? 1 : ((v[i] < -thr) ? -1 : 0);
        lo |= (q & 0xff) << (8 * i);
    }
#pragma unroll
    for (int i = 0; i < 4; ++i) {
        int q = (v[4 + i] > thr) ? 1 : ((v[4 + i] < -thr) ? -1 : 0);
        hi |= (q & 0xff) << (8 * i);
    }
    *(int2*)(wq + base) = make_int2(lo, hi);
}

// ---------------------------------------------------------------------------
// Pass 2: per-row activation quantization. One block per row of 2048.
// ---------------------------------------------------------------------------
__global__ __launch_bounds__(256) void quant_x_kernel(const float* __restrict__ x,
                                                      signed char* __restrict__ xq,
                                                      float* __restrict__ scales) {
    __shared__ float red[256];
    const int row = blockIdx.x;
    const int t = threadIdx.x;
    const float* xr = x + (size_t)row * K_DIM;
    const float4 a = ((const float4*)xr)[t * 2];
    const float4 b = ((const float4*)xr)[t * 2 + 1];
    float v[8] = {a.x, a.y, a.z, a.w, b.x, b.y, b.z, b.w};
    float mx = 0.0f;
#pragma unroll
    for (int i = 0; i < 8; ++i) mx = fmaxf(mx, fabsf(v[i]));
    red[t] = mx;
    __syncthreads();
    for (int off = 128; off > 0; off >>= 1) {
        if (t < off) red[t] = fmaxf(red[t], red[t + off]);
        __syncthreads();
    }
    const float scale = fmaxf(red[0], 1e-5f);
    if (t == 0) scales[row] = scale;
    const float inv = 127.0f / scale;
    int lo = 0, hi = 0;
#pragma unroll
    for (int i = 0; i < 4; ++i) {
        float r = __builtin_rintf(v[i] * inv);            // round-half-even, matches jnp.round
        r = fminf(fmaxf(r, -127.0f), 127.0f);
        lo |= ((int)r & 0xff) << (8 * i);
    }
#pragma unroll
    for (int i = 0; i < 4; ++i) {
        float r = __builtin_rintf(v[4 + i] * inv);
        r = fminf(fmaxf(r, -127.0f), 127.0f);
        hi |= ((int)r & 0xff) << (8 * i);
    }
    *(int2*)(xq + (size_t)row * K_DIM + t * 8) = make_int2(lo, hi);
}

// ---------------------------------------------------------------------------
// 8-bit WMMA operand load: ISA A-matrix 16x64 layout (B is symmetric over N).
// Lane L: row = L&15, K-half offset = (L>>4)*8.
// VGPR pairs: K {0,16,32,48}+koff .. +7
// ---------------------------------------------------------------------------
__device__ __forceinline__ v8i load_q8_tile(const signed char* __restrict__ base,
                                            int lane, int k0) {
    const int r = lane & 15;
    const int koff = (lane >> 4) << 3;
    const signed char* p = base + (size_t)r * K_DIM + k0 + koff;
    const v2i t0 = *(const v2i*)(p);
    const v2i t1 = *(const v2i*)(p + 16);
    const v2i t2 = *(const v2i*)(p + 32);
    const v2i t3 = *(const v2i*)(p + 48);
    v8i out;
    out[0] = t0[0]; out[1] = t0[1];
    out[2] = t1[0]; out[3] = t1[1];
    out[4] = t2[0]; out[5] = t2[1];
    out[6] = t3[0]; out[7] = t3[1];
    return out;
}

// ---------------------------------------------------------------------------
// Pass 3: int8 x ternary-int8 GEMM with V_WMMA_I32_16X16X64_IU8.
// Block = 256 threads = 8 waves arranged 2(M) x 4(N).
// Wave tile = 32x64 (2x4 accumulators). Block tile = 64x256.
// Grid = (N/256, M/64) = (32, 128).
// ---------------------------------------------------------------------------
__global__ __launch_bounds__(256) void gemm_kernel(const signed char* __restrict__ xq,
                                                   const signed char* __restrict__ wq,
                                                   const float* __restrict__ scales,
                                                   const float* __restrict__ wsum,
                                                   float* __restrict__ out) {
    const int lane = threadIdx.x & 31;
    const int wave = threadIdx.x >> 5;
    const int wm = wave >> 2;           // 0..1
    const int wn = wave & 3;            // 0..3
    const int tile_m = blockIdx.y * 64 + wm * 32;
    const int tile_n = blockIdx.x * 256 + wn * 64;

    v8i c[2][4] = {};

    const signed char* abase = xq + (size_t)tile_m * K_DIM;
    const signed char* bbase = wq + (size_t)tile_n * K_DIM;

    for (int k0 = 0; k0 < K_DIM; k0 += 64) {
        v8i a[2], b[4];
#pragma unroll
        for (int i = 0; i < 2; ++i)
            a[i] = load_q8_tile(abase + (size_t)i * 16 * K_DIM, lane, k0);
#pragma unroll
        for (int j = 0; j < 4; ++j)
            b[j] = load_q8_tile(bbase + (size_t)j * 16 * K_DIM, lane, k0);
#pragma unroll
        for (int i = 0; i < 2; ++i)
#pragma unroll
            for (int j = 0; j < 4; ++j)
                c[i][j] = __builtin_amdgcn_wmma_i32_16x16x64_iu8(
                    /*sgn_a=*/true, a[i], /*sgn_b=*/true, b[j], c[i][j],
                    /*reuse_a=*/false, /*reuse_b=*/false);
    }

    // Epilogue: out = (max(acc,0) * w_scale / scale[m])^2
    // C/D layout: VGPR r -> M = r + 8*(lane>=16), N = lane & 15.
    const float wscale = wsum[0] * (1.0f / 16777216.0f);
    const int nn = lane & 15;
    const int mh = (lane >> 4) * 8;
#pragma unroll
    for (int i = 0; i < 2; ++i) {
#pragma unroll
        for (int r = 0; r < 8; ++r) {
            const int m = tile_m + i * 16 + mh + r;
            const float f = wscale / scales[m];
            float* orow = out + (size_t)m * N_DIM + tile_n + nn;
#pragma unroll
            for (int j = 0; j < 4; ++j) {
                float v = fmaxf((float)c[i][j][r], 0.0f) * f;
                orow[j * 16] = v * v;
            }
        }
    }
}

extern "C" void kernel_launch(void* const* d_in, const int* in_sizes, int n_in,
                              void* d_out, int out_size, void* d_ws, size_t ws_size,
                              hipStream_t stream) {
    const float* x  = (const float*)d_in[0];
    const float* wt = (const float*)d_in[1];
    float* out = (float*)d_out;

    char* ws = (char*)d_ws;
    float* wsum         = (float*)ws;
    float* scales       = (float*)(ws + 256);
    signed char* xq     = (signed char*)(ws + 65536);
    signed char* wq     = (signed char*)(ws + 65536 + (16u << 20));

    hipMemsetAsync(wsum, 0, sizeof(float), stream);

    wsum_kernel<<<2048, 256, 0, stream>>>(wt, wsum);
    quant_w_kernel<<<W_ELEMS / (256 * 8), 256, 0, stream>>>(wt, wsum, wq);
    quant_x_kernel<<<M_DIM, 256, 0, stream>>>(x, xq, scales);

    dim3 grid(N_DIM / 256, M_DIM / 64);
    gemm_kernel<<<grid, 256, 0, stream>>>(xq, wq, scales, wsum, out);
}
